// SelfAttention_10264971837480
// MI455X (gfx1250) — matrix operand order repaired and checked by reading, not compiled, not run on hardware
//
#include <hip/hip_runtime.h>
#include <hip/hip_bf16.h>
#include <stdint.h>

// Problem constants (from reference)
#define NB     2
#define SEQ    2048
#define EMBED  1024
#define HEADS  16
#define HDIM   64
#define NTILES (SEQ / 64)

typedef __attribute__((ext_vector_type(16))) __bf16 v16bf;
typedef __attribute__((ext_vector_type(8)))  __bf16 v8bf;
typedef __attribute__((ext_vector_type(8)))  float  v8f;

union ABfrag { v16bf v; v8bf h[2]; };

__device__ __forceinline__ __bf16 f2bf(float f) {
  uint32_t u = __builtin_bit_cast(uint32_t, f);
  uint32_t r = u + 0x7FFFu + ((u >> 16) & 1u);   // round-to-nearest-even
  uint16_t hh = (uint16_t)(r >> 16);
  return __builtin_bit_cast(__bf16, hh);
}

// Per-lane async DMA: 16 bytes global -> LDS, tracked by ASYNCcnt.
__device__ __forceinline__ void async_b128(uint32_t lds_addr, const void* gptr) {
  asm volatile("global_load_async_to_lds_b128 %0, %1, off"
               :: "v"(lds_addr), "v"((uint64_t)(uintptr_t)gptr)
               : "memory");
}

// ---------------------------------------------------------------------------
// Prologue: f32 -> bf16.  Q/K head-major [n][h][l][64]; V transposed
// [n][h][d][l] so attention staging is pure contiguous 16B async chunks.
// ---------------------------------------------------------------------------
__global__ void cvt_qk_kernel(const float* __restrict__ q,
                              const float* __restrict__ k,
                              __bf16* __restrict__ qh,
                              __bf16* __restrict__ kh) {
  size_t idx = (size_t)blockIdx.x * blockDim.x + threadIdx.x;
  size_t total = (size_t)NB * SEQ * EMBED;
  if (idx >= total) return;
  int e = (int)(idx % EMBED);
  size_t nl = idx / EMBED;
  int l = (int)(nl % SEQ);
  int n = (int)(nl / SEQ);
  int h = e / HDIM, d = e % HDIM;
  size_t dst = (((size_t)(n * HEADS + h) * SEQ) + l) * HDIM + d;
  qh[dst] = f2bf(q[idx]);
  kh[dst] = f2bf(k[idx]);
}

__global__ void cvt_v_kernel(const float* __restrict__ v,
                             __bf16* __restrict__ vt) {
  // thread -> (n, h, l0, d); reads coalesced across d, writes 16B chunks.
  size_t idx = (size_t)blockIdx.x * blockDim.x + threadIdx.x;
  size_t total = (size_t)NB * HEADS * HDIM * (SEQ / 8);
  if (idx >= total) return;
  int d = (int)(idx % HDIM);
  size_t rest = idx / HDIM;
  int l0 = (int)(rest % (SEQ / 8)) * 8;
  size_t nh = rest / (SEQ / 8);
  int h = (int)(nh % HEADS);
  int n = (int)(nh / HEADS);
  const float* src = v + ((size_t)n * SEQ + l0) * EMBED + h * HDIM + d;
  v8bf out;
#pragma unroll
  for (int i = 0; i < 8; ++i) out[i] = f2bf(src[(size_t)i * EMBED]);
  *(v8bf*)(vt + ((size_t)(n * HEADS + h) * HDIM + d) * SEQ + l0) = out;
}

__global__ void cvt_wo_kernel(const float* __restrict__ wo,
                              __bf16* __restrict__ wob) {
  size_t idx = (size_t)blockIdx.x * blockDim.x + threadIdx.x;
  if (idx < (size_t)EMBED * EMBED) wob[idx] = f2bf(wo[idx]);
}

// ---------------------------------------------------------------------------
// Flash-attention kernel.
// Grid: (NB*HEADS, SEQ/128).  Block: 256 threads = 8 waves (wave32).
// Double-buffered async global->LDS staging of 64-key K/V tiles;
// all matmuls via v_wmma_f32_16x16x32_bf16.
// ---------------------------------------------------------------------------
__global__ void __launch_bounds__(256, 2)
attn_kernel(const __bf16* __restrict__ Qh, const __bf16* __restrict__ Kh,
            const __bf16* __restrict__ Vt, const int* __restrict__ mask,
            __bf16* __restrict__ Obuf) {
  __shared__ __bf16 Klds[2][64 * 64];    // [key][d]   2 x 8 KB
  __shared__ __bf16 Vtlds[2][64 * 64];   // [d][key]   2 x 8 KB
  __shared__ __bf16 Plds[8][16 * 64];    // per-wave P tile, 16 KB

  const int tid    = threadIdx.x;
  const int wid    = tid >> 5;
  const int lane   = tid & 31;
  const int laneLo = lane & 15;
  const int laneHi = lane >> 4;

  const int n = blockIdx.x / HEADS;
  const int h = blockIdx.x % HEADS;
  const int qBase = blockIdx.y * 128 + wid * 16;

  const __bf16* Qhead = Qh + (size_t)(n * HEADS + h) * SEQ * HDIM;
  const __bf16* Khead = Kh + (size_t)(n * HEADS + h) * SEQ * HDIM;
  const __bf16* Vthd  = Vt + (size_t)(n * HEADS + h) * HDIM * SEQ;
  const int* maskN = mask + (size_t)n * SEQ * SEQ;

  // Issue one 64x64 bf16 K tile + V tile copy: 4 async b128 per thread.
  auto issue_tile = [&](int kb, int buf) {
    const int keyBase = kb * 64;
#pragma unroll
    for (int c = 0; c < 2; ++c) {
      const int chunk = tid + c * 256;   // 0..511 chunks of 8 bf16
      // K tile is contiguous in global ([key][d] rows back-to-back)
      async_b128((uint32_t)(uintptr_t)&Klds[buf][chunk * 8],
                 Khead + (size_t)keyBase * HDIM + chunk * 8);
      // V tile: 64 d-rows of 64 keys; each row contiguous in global Vt
      const int d = chunk >> 3, part = chunk & 7;
      async_b128((uint32_t)(uintptr_t)&Vtlds[buf][d * 64 + part * 8],
                 Vthd + (size_t)d * SEQ + keyBase + part * 8);
    }
  };

  // Q A-fragments (16 x 64 as two 16x32 K-slices); A layout:
// lane&15 = row M, element e -> k = kBase + 16*laneHi + e
  ABfrag aQ[2];
  {
    const __bf16* qp = Qhead + (size_t)(qBase + laneLo) * HDIM;
#pragma unroll
    for (int t = 0; t < 2; ++t) {
      aQ[t].h[0] = *(const v8bf*)(qp + 32 * t + 16 * laneHi);
      aQ[t].h[1] = *(const v8bf*)(qp + 32 * t + 8 + 16 * laneHi);
    }
  }

  v8f O[4];
#pragma unroll
  for (int t = 0; t < 4; ++t) O[t] = {};
  float mrow[8], lrow[8];
#pragma unroll
  for (int r = 0; r < 8; ++r) { mrow[r] = -3.0e38f; lrow[r] = 0.0f; }

  const float scale = 0.03125f;  // 1/sqrt(EMBED) = 1/32

  issue_tile(0, 0);

  for (int kb = 0; kb < NTILES; ++kb) {
    const int buf = kb & 1;
    if (kb + 1 < NTILES) {
      issue_tile(kb + 1, buf ^ 1);
      // 8 outstanding now; drain down to the 4 just issued -> tile kb done
      asm volatile("s_wait_asynccnt 0x4" ::: "memory");
    } else {
      asm volatile("s_wait_asynccnt 0x0" ::: "memory");
    }
    __syncthreads();   // all waves' tile-kb copies visible

    const int keyBase = kb * 64;

    // ---- S = Q @ K^T : 16 x 64 in 4 C-fragments ----
    v8f S[4];
#pragma unroll
    for (int t = 0; t < 4; ++t) {
      // B layout: lane&15 = column (key), element e -> k = 16*laneHi + e
      const __bf16* kp = &Klds[buf][(t * 16 + laneLo) * HDIM];
      ABfrag b0, b1;
      b0.v = *(const v16bf*)(kp + 16 * laneHi);       // d = 0..31
      b1.v = *(const v16bf*)(kp + 32 + 16 * laneHi);  // d = 32..63
      v8f acc = {};
      acc = __builtin_amdgcn_wmma_f32_16x16x32_bf16(false, aQ[0].v, false, b0.v,
                                                    (short)0, acc, false, false);
      acc = __builtin_amdgcn_wmma_f32_16x16x32_bf16(false, aQ[1].v, false, b1.v,
                                                    (short)0, acc, false, false);
      S[t] = acc;
    }

    // ---- mask + scale + per-row block max ----
    float bmax[8];
#pragma unroll
    for (int r = 0; r < 8; ++r) bmax[r] = -3.0e38f;
#pragma unroll
    for (int t = 0; t < 4; ++t) {
      const int kcol = keyBase + t * 16 + laneLo;
#pragma unroll
      for (int r = 0; r < 8; ++r) {
        const int qrow = qBase + r + 8 * laneHi;
        float s = S[t][r];
        if (maskN[(size_t)qrow * SEQ + kcol] == 0) s = -1.0e20f;
        s *= scale;
        S[t][r] = s;
        bmax[r] = fmaxf(bmax[r], s);
      }
    }
    // C layout: row = r + 8*laneHi, cols across the 16 lanes of each half
#pragma unroll
    for (int r = 0; r < 8; ++r) {
      float v = bmax[r];
      v = fmaxf(v, __shfl_xor(v, 1, 32));
      v = fmaxf(v, __shfl_xor(v, 2, 32));
      v = fmaxf(v, __shfl_xor(v, 4, 32));
      v = fmaxf(v, __shfl_xor(v, 8, 32));
      bmax[r] = v;
    }
    float alpha[8], rsum[8];
#pragma unroll
    for (int r = 0; r < 8; ++r) {
      float mnew = fmaxf(mrow[r], bmax[r]);
      alpha[r] = __expf(mrow[r] - mnew);
      mrow[r] = mnew;
      rsum[r] = 0.0f;
    }
    // ---- P = exp(S - m) -> per-wave LDS tile (bf16), accumulate row sums ----
#pragma unroll
    for (int t = 0; t < 4; ++t) {
#pragma unroll
      for (int r = 0; r < 8; ++r) {
        float p = __expf(S[t][r] - mrow[r]);
        rsum[r] += p;
        Plds[wid][(r + 8 * laneHi) * 64 + t * 16 + laneLo] = f2bf(p);
      }
    }
#pragma unroll
    for (int r = 0; r < 8; ++r) {
      float v = rsum[r];
      v += __shfl_xor(v, 1, 32);
      v += __shfl_xor(v, 2, 32);
      v += __shfl_xor(v, 4, 32);
      v += __shfl_xor(v, 8, 32);
      lrow[r] = lrow[r] * alpha[r] + v;
    }
#pragma unroll
    for (int t = 0; t < 4; ++t)
#pragma unroll
      for (int r = 0; r < 8; ++r) O[t][r] *= alpha[r];

    // ---- reload P as A-fragments (LDS ops in-order within a wave) ----
    ABfrag aP[2];
    {
      const __bf16* pp = &Plds[wid][laneLo * 64];
#pragma unroll
      for (int u = 0; u < 2; ++u) {
        aP[u].h[0] = *(const v8bf*)(pp + 32 * u + 16 * laneHi);
        aP[u].h[1] = *(const v8bf*)(pp + 32 * u + 8 + 16 * laneHi);
      }
    }
    // ---- O += P @ V (V transposed in LDS -> contiguous B frags) ----
#pragma unroll
    for (int t = 0; t < 4; ++t) {
      const __bf16* vp = &Vtlds[buf][(t * 16 + laneLo) * 64];
      ABfrag bv0, bv1;
      bv0.v = *(const v16bf*)(vp + 16 * laneHi);       // keys 0..31
      bv1.v = *(const v16bf*)(vp + 32 + 16 * laneHi);  // keys 32..63
      O[t] = __builtin_amdgcn_wmma_f32_16x16x32_bf16(false, aP[0].v, false, bv0.v,
                                                     (short)0, O[t], false, false);
      O[t] = __builtin_amdgcn_wmma_f32_16x16x32_bf16(false, aP[1].v, false, bv1.v,
                                                     (short)0, O[t], false, false);
    }
    __syncthreads();   // readers done before next iteration overwrites buf^1
  }

  // ---- epilogue: normalize and write bf16 attention output [n][l][e] ----
#pragma unroll
  for (int t = 0; t < 4; ++t) {
    const int ecol = h * HDIM + t * 16 + laneLo;
#pragma unroll
    for (int r = 0; r < 8; ++r) {
      const int qrow = qBase + r + 8 * laneHi;
      float o = O[t][r] / lrow[r];
      Obuf[((size_t)n * SEQ + qrow) * EMBED + ecol] = f2bf(o);
    }
  }
}

// ---------------------------------------------------------------------------
// Output projection: Y[M=4096, E] = X_bf16 @ Wo^T + bo   (f32 out)
// Grid: (M/128, E/64), 8 waves x 16 rows x 64 cols per block.
// ---------------------------------------------------------------------------
__global__ void __launch_bounds__(256, 2)
proj_kernel(const __bf16* __restrict__ X, const __bf16* __restrict__ W,
            const float* __restrict__ bias, float* __restrict__ Y) {
  const int tid    = threadIdx.x;
  const int wid    = tid >> 5;
  const int lane   = tid & 31;
  const int laneLo = lane & 15;
  const int laneHi = lane >> 4;
  const int rBase = blockIdx.x * 128 + wid * 16;
  const int cBase = blockIdx.y * 64;

  v8f C[4];
#pragma unroll
  for (int t = 0; t < 4; ++t) C[t] = {};

  const __bf16* xrow = X + (size_t)(rBase + laneLo) * EMBED;

  for (int kk = 0; kk < EMBED; kk += 32) {
    ABfrag a;
    a.h[0] = *(const v8bf*)(xrow + kk + 16 * laneHi);
    a.h[1] = *(const v8bf*)(xrow + kk + 8 + 16 * laneHi);
#pragma unroll
    for (int t = 0; t < 4; ++t) {
      ABfrag b;
      b.v = *(const v16bf*)(W + (size_t)(cBase + t * 16 + laneLo) * EMBED +
                            kk + 16 * laneHi);
      C[t] = __builtin_amdgcn_wmma_f32_16x16x32_bf16(false, a.v, false, b.v,
                                                     (short)0, C[t], false, false);
    }
  }

#pragma unroll
  for (int t = 0; t < 4; ++t) {
    const int col = cBase + t * 16 + laneLo;
    const float b = bias[col];
#pragma unroll
    for (int r = 0; r < 8; ++r)
      Y[(size_t)(rBase + r + 8 * laneHi) * EMBED + col] = C[t][r] + b;
  }
}

// ---------------------------------------------------------------------------
// Launch.  Workspace layout (bf16):
//   [0,8MB)   Qh head-major   [8,16MB) Kh   [16,24MB) Vt (transposed)
//   [24,32MB) attention output (pre-projection)
//   [32,34MB) Wo bf16
// ---------------------------------------------------------------------------
extern "C" void kernel_launch(void* const* d_in, const int* in_sizes, int n_in,
                              void* d_out, int out_size, void* d_ws, size_t ws_size,
                              hipStream_t stream) {
  const float* values = (const float*)d_in[0];
  const float* keysp  = (const float*)d_in[1];
  const float* query  = (const float*)d_in[2];
  const int*   mask   = (const int*)d_in[3];
  const float* Wo     = (const float*)d_in[4];
  const float* bo     = (const float*)d_in[5];
  float* out = (float*)d_out;

  char* ws = (char*)d_ws;
  const size_t headBytes = (size_t)NB * HEADS * SEQ * HDIM * sizeof(__bf16); // 8 MB
  __bf16* Qh  = (__bf16*)(ws);
  __bf16* Kh  = (__bf16*)(ws + headBytes);
  __bf16* Vt  = (__bf16*)(ws + 2 * headBytes);
  __bf16* Ob  = (__bf16*)(ws + 3 * headBytes);
  __bf16* WoB = (__bf16*)(ws + 4 * headBytes);

  {
    size_t total = (size_t)NB * SEQ * EMBED;
    cvt_qk_kernel<<<(int)((total + 255) / 256), 256, 0, stream>>>(query, keysp,
                                                                  Qh, Kh);
  }
  {
    size_t total = (size_t)NB * HEADS * HDIM * (SEQ / 8);
    cvt_v_kernel<<<(int)((total + 255) / 256), 256, 0, stream>>>(values, Vt);
  }
  {
    size_t total = (size_t)EMBED * EMBED;
    cvt_wo_kernel<<<(int)((total + 255) / 256), 256, 0, stream>>>(Wo, WoB);
  }
  attn_kernel<<<dim3(NB * HEADS, SEQ / 128), 256, 0, stream>>>(Qh, Kh, Vt, mask, Ob);
  proj_kernel<<<dim3(NB * SEQ / 128, EMBED / 64), 256, 0, stream>>>(Ob, WoB, bo, out);
}
